// ZeroDivMat_70961449664818
// MI455X (gfx1250) — matrix operand order repaired and checked by reading, not compiled, not run on hardware
//
#include <hip/hip_runtime.h>
#include <math.h>

#define NDIM   8
#define NPAIR  28
#define NH     128
#define MTILE  128   // batch rows per block
#define WROWS  16    // batch rows per wave
#define SH_STRIDE 132
#define SW_STRIDE 132

typedef __attribute__((ext_vector_type(16))) __bf16 bf16x16;
typedef __attribute__((ext_vector_type(8)))  __bf16 bf16x8;
typedef __attribute__((ext_vector_type(8)))  float  f32x8;

struct alignas(8) u32x2 { unsigned x, y; };

// single f32 -> bf16 (round-to-nearest, ties away): 2 VALU
__device__ __forceinline__ unsigned short f2bf(float f) {
  return (unsigned short)((__float_as_uint(f) + 0x8000u) >> 16);
}
__device__ __forceinline__ float bf2f(unsigned short h) {
  return __uint_as_float(((unsigned)h) << 16);
}
// two f32 -> packed bf16x2 (a in [15:0], b in [31:16]): 2 adds + 1 v_perm_b32
__device__ __forceinline__ unsigned f2bf_pk(float a, float b) {
  const unsigned ua = __float_as_uint(a) + 0x8000u;
  const unsigned ub = __float_as_uint(b) + 0x8000u;
  // sel bytes: 0x02,0x03 -> S1(=ua) hi16 ; 0x06,0x07 -> S0(=ub) hi16
  return __builtin_amdgcn_perm(ub, ua, 0x07060302u);
}

// fast softplus: raw v_exp_f32/v_log_f32 (arg of log is in (1,2] -> no denorm path)
__device__ __forceinline__ float softplus_f(float v) {
#if __has_builtin(__builtin_amdgcn_exp2f) && __has_builtin(__builtin_amdgcn_logf)
  const float t  = __builtin_amdgcn_exp2f(fabsf(v) * -1.442695040888963f);
  const float lg = __builtin_amdgcn_logf(1.f + t);          // log2(1+t)
  return fmaf(0.6931471805599453f, lg, fmaxf(v, 0.f));
#else
  return fmaxf(v, 0.f) + __logf(1.f + __expf(-fabsf(v)));
#endif
}

__global__ __launch_bounds__(256)
void pairnet_wmma_kernel(const float* __restrict__ x,
                         const float* __restrict__ W1,
                         const float* __restrict__ b1,
                         const float* __restrict__ Wh,
                         const float* __restrict__ bh,
                         const float* __restrict__ Wout,
                         const float* __restrict__ bout,
                         const int*   __restrict__ rel,
                         const int*   __restrict__ iidx,
                         const int*   __restrict__ jidx,
                         float*       __restrict__ out)
{
  __shared__ unsigned short sW[NH * SW_STRIDE];     // current layer weights, bf16
  __shared__ unsigned short sH[MTILE * SH_STRIDE];  // activations, bf16 (per-wave 16-row slices)

  const int tid   = threadIdx.x;
  const int wid   = tid >> 5;
  const int lane  = tid & 31;
  const int lrow  = lane & 15;
  const int lhalf = lane >> 4;
  const int p     = blockIdx.y;
  const int row0  = blockIdx.x * MTILE;

  // ---- stage W1 (6x128) bf16 into sW rows 0..5; zero rows 6..31 -----------
  {
    const float* w1p = W1 + (long)p * 6 * NH;
    for (int idx = tid; idx < 6 * NH; idx += 256) {
      const int d = idx >> 7, c = idx & 127;
      sW[d * SW_STRIDE + c] = f2bf(w1p[idx]);
    }
    for (int idx = tid; idx < 26 * SW_STRIDE; idx += 256)
      sW[6 * SW_STRIDE + idx] = 0;                  // K=6..31 must be exact 0
  }
  __syncthreads();

  // ---- layer 1 via WMMA: one 16x16x32 per N-tile (K=0..5 live) ------------
  {
    const long b = (long)row0 + wid * WROWS + lrow;
    union { bf16x16 v; unsigned short u[16]; } a1;
#pragma unroll
    for (int e = 0; e < 16; ++e) a1.u[e] = 0;
    if (lhalf == 0) {                               // lanes 0-15: K = e (e<8)
#pragma unroll
      for (int d = 0; d < 6; ++d)
        a1.u[d] = f2bf(x[b * NDIM + rel[p * 6 + d]]);
    }
    const float* b1p = b1 + (long)p * NH;
    float bias[8];
#pragma unroll
    for (int nt = 0; nt < 8; ++nt) bias[nt] = b1p[nt * 16 + lrow];

    for (int nt = 0; nt < 8; ++nt) {
      const float bv = bias[nt];                    // D layout: N = lane%16
      f32x8 acc = {bv, bv, bv, bv, bv, bv, bv, bv};
      union { bf16x16 v; unsigned short u[16]; } bq;
#pragma unroll
      for (int s = 0; s < 16; ++s)
        bq.u[s] = sW[(lhalf * 8 + (s & 7) + (s >> 3) * 16) * SW_STRIDE + nt * 16 + lrow];
      bf16x16 Bf = bq.v;
      acc = __builtin_amdgcn_wmma_f32_16x16x32_bf16(
                false, a1.v, false, Bf, (short)0, acc, false, false);
#pragma unroll
      for (int r = 0; r < 8; r += 2) {
        const unsigned pk = f2bf_pk(softplus_f(acc[r]), softplus_f(acc[r + 1]));
        const int base = (wid * WROWS + r + lhalf * 8) * SH_STRIDE + nt * 16 + lrow;
        sH[base]             = (unsigned short)pk;          // ds_store_b16
        sH[base + SH_STRIDE] = (unsigned short)(pk >> 16);  // ds_store_b16_d16_hi
      }
    }
  }
  __syncthreads();   // all waves done reading sW (W1) before Wh overwrite

  // ---- hidden layers: h = softplus(h @ Wh[l] + bh[l]) ---------------------
  for (int l = 0; l < 2; ++l) {
    // cooperative load + f32->bf16 convert of Wh[l][p] (128x128) into LDS
    const float* wl = Wh + ((long)l * NPAIR + p) * NH * NH;
#pragma unroll
    for (int i = 0; i < 16; ++i) {
      const int e = (i * 256 + tid) * 4;            // float index, 16B aligned
      const int r = e >> 7;
      const int c = e & 127;
      const float4 w = *(const float4*)(wl + e);
      u32x2 h2;
      h2.x = f2bf_pk(w.x, w.y);
      h2.y = f2bf_pk(w.z, w.w);
      *(u32x2*)&sW[r * SW_STRIDE + c] = h2;
    }
    if (l == 0) {                                   // warm L2/L0 for next layer
      const float* nw = Wh + ((long)NPAIR + p) * NH * NH;
      __builtin_prefetch(nw + tid * 64, 0, 1);
    }
    __syncthreads();

    // A fragments: 16x128 bf16 activations -> 4 chunks of 16x32
    // ISA layout: lane M = lane%16; elem e<8 -> K = e + 8*(lane/16); e>=8 -> +16
    bf16x16 A[4];
    const int arow = (wid * WROWS + lrow) * SH_STRIDE;
#pragma unroll
    for (int kc = 0; kc < 4; ++kc) {
      bf16x8 lo = *(const bf16x8*)&sH[arow + kc * 32 + lhalf * 8];
      bf16x8 hi = *(const bf16x8*)&sH[arow + kc * 32 + lhalf * 8 + 16];
      A[kc] = __builtin_shufflevector(lo, hi,
              0,1,2,3,4,5,6,7,8,9,10,11,12,13,14,15);
    }

    const float* blp = bh + ((long)l * NPAIR + p) * NH;
    float bias[8];
#pragma unroll
    for (int nt = 0; nt < 8; ++nt) bias[nt] = blp[nt * 16 + lrow];

    for (int nt = 0; nt < 8; ++nt) {
      const float bv = bias[nt];                    // D layout: N = lane%16
      f32x8 acc = {bv, bv, bv, bv, bv, bv, bv, bv};
#pragma unroll
      for (int kc = 0; kc < 4; ++kc) {
        // B fragment: lane holds column N = nt*16 + lane%16; elem e<8 -> K = kc*32 + e + 8*(lane/16); e>=8 -> +16
        union { bf16x16 v; unsigned short u[16]; } bq;
#pragma unroll
        for (int s = 0; s < 16; ++s)
          bq.u[s] = sW[(kc * 32 + lhalf * 8 + (s & 7) + (s >> 3) * 16) * SW_STRIDE + nt * 16 + lrow];
        bf16x16 Bf = bq.v;
        acc = __builtin_amdgcn_wmma_f32_16x16x32_bf16(
                  false, A[kc], false, Bf, (short)0, acc, false, false);
      }
      // D layout: M = r + 8*(lane/16), N = nt*16 + lane%16
#pragma unroll
      for (int r = 0; r < 8; r += 2) {
        const unsigned pk = f2bf_pk(softplus_f(acc[r]), softplus_f(acc[r + 1]));
        const int base = (wid * WROWS + r + lhalf * 8) * SH_STRIDE + nt * 16 + lrow;
        sH[base]             = (unsigned short)pk;
        sH[base + SH_STRIDE] = (unsigned short)(pk >> 16);
      }
    }
    __syncthreads();
  }

  // ---- output: vals = h @ Wout + bout, scatter antisymmetric --------------
  {
    const int ii = iidx[p], jj = jidx[p];
    const float* wo = Wout + (long)p * NH + lhalf * 64;
    const int rowbase = (wid * WROWS + lrow) * SH_STRIDE + lhalf * 64;
    float part = 0.f;
#pragma unroll
    for (int c8 = 0; c8 < 8; ++c8) {
      union { bf16x8 v; unsigned short u[8]; } hv;
      hv.v = *(const bf16x8*)&sH[rowbase + c8 * 8];
      const float4 w0 = *(const float4*)(wo + c8 * 8);
      const float4 w1 = *(const float4*)(wo + c8 * 8 + 4);
      part = fmaf(bf2f(hv.u[0]), w0.x, part);
      part = fmaf(bf2f(hv.u[1]), w0.y, part);
      part = fmaf(bf2f(hv.u[2]), w0.z, part);
      part = fmaf(bf2f(hv.u[3]), w0.w, part);
      part = fmaf(bf2f(hv.u[4]), w1.x, part);
      part = fmaf(bf2f(hv.u[5]), w1.y, part);
      part = fmaf(bf2f(hv.u[6]), w1.z, part);
      part = fmaf(bf2f(hv.u[7]), w1.w, part);
    }
    part += __shfl_xor(part, 16, 32);               // combine the two halves
    if (lane < 16) {
      const long  b   = (long)row0 + wid * WROWS + lrow;
      const float val = part + bout[p];
      out[b * 64 + ii * 8 + jj] =  val;
      out[b * 64 + jj * 8 + ii] = -val;
    }
  }
}

extern "C" void kernel_launch(void* const* d_in, const int* in_sizes, int n_in,
                              void* d_out, int out_size, void* d_ws, size_t ws_size,
                              hipStream_t stream) {
  const float* x    = (const float*)d_in[0];
  const float* W1   = (const float*)d_in[1];
  const float* b1   = (const float*)d_in[2];
  const float* Wh   = (const float*)d_in[3];
  const float* bh   = (const float*)d_in[4];
  const float* Wout = (const float*)d_in[5];
  const float* bout = (const float*)d_in[6];
  const int*   rel  = (const int*)d_in[7];
  const int*   ii   = (const int*)d_in[8];
  const int*   jj   = (const int*)d_in[9];
  float* out = (float*)d_out;

  const int batch = in_sizes[0] / NDIM;             // 65536

  // zero output (diagonal entries are never written by the kernel)
  hipMemsetAsync(d_out, 0, (size_t)out_size * sizeof(float), stream);

  dim3 grid(batch / MTILE, NPAIR, 1);
  dim3 block(256, 1, 1);
  pairnet_wmma_kernel<<<grid, block, 0, stream>>>(
      x, W1, b1, Wh, bh, Wout, bout, rel, ii, jj, out);
}
